// MultiGraph_16252156248539
// MI455X (gfx1250) — matrix operand order repaired and checked by reading, not compiled, not run on hardware
//
#include <hip/hip_runtime.h>

// ---------------------------------------------------------------------------
// MultiGraph hetero-GNN forward for MI455X (gfx1250, wave32, WMMA bf16).
// ---------------------------------------------------------------------------

constexpr int HD = 128;   // hidden dim (== GEMM K for every matmul here)
constexpr int KDIM = 128;
constexpr int NPAP = 4096;
constexpr int NAUT = 2048;

typedef __attribute__((ext_vector_type(16))) __bf16 v16bf;
typedef __attribute__((ext_vector_type(8)))  float  v8f;

static __device__ __forceinline__ unsigned short f2bf_bits(float f) {
  unsigned u = __builtin_bit_cast(unsigned, f);
  unsigned r = u + 0x7FFFu + ((u >> 16) & 1u);   // round-to-nearest-even
  return (unsigned short)(r >> 16);
}
static __device__ __forceinline__ __bf16 bits2bf(unsigned short h) {
  return __builtin_bit_cast(__bf16, h);
}

// ---------------------------------------------------------------------------
// WMMA GEMM: C[M,N] = A[M,K=128] * B[128,N], fp32 in/out, bf16 mul, f32 accum.
// grid = (N/32, M/64), block = 128 (4 waves).
// Each wave: 16x32 output tile = 2 accumulators sharing one A fragment
// (doubles WMMA per A load; the WMMA pair is hazard-free back-to-back).
// B tile (32k x 32n) staged in LDS transposed as bf16, column-per-lane read.
// K unrolled at compile time -> 8 static v_wmma per kernel.
// ---------------------------------------------------------------------------
__global__ __launch_bounds__(128)
void gemm_bf16_wmma(const float* __restrict__ A, const float* __restrict__ B,
                    float* __restrict__ C, int lda, int ldb, int ldc)
{
  __shared__ unsigned short ldsB[32][34];   // [n][k], padded

  const int lane  = threadIdx.x & 31;
  const int wave  = threadIdx.x >> 5;
  const int col0  = blockIdx.x * 32;
  const int mrow0 = (blockIdx.y * 4 + wave) * 16;

  v8f acc0 = {0.f, 0.f, 0.f, 0.f, 0.f, 0.f, 0.f, 0.f};
  v8f acc1 = {0.f, 0.f, 0.f, 0.f, 0.f, 0.f, 0.f, 0.f};

  const unsigned arow  = (unsigned)(mrow0 + (lane & 15));
  const int      ahalf = (lane < 16) ? 0 : 16;   // K sub-block per lane half
  const int      bcol  = lane & 15;
  const int      bko   = (lane < 16) ? 0 : 16;

  #pragma unroll
  for (int kb = 0; kb < KDIM; kb += 32) {
    __syncthreads();
    // stage B[kb..kb+31][col0..col0+31] -> ldsB[n][k] (1024 elems, 8/thread,
    // consecutive threads hit consecutive columns -> coalesced)
    #pragma unroll
    for (int i = 0; i < 8; ++i) {
      int idx = (int)threadIdx.x + i * 128;
      int kk = idx >> 5;
      int nn = idx & 31;
      ldsB[nn][kk] = f2bf_bits(B[(unsigned)(kb + kk) * (unsigned)ldb +
                                 (unsigned)(col0 + nn)]);
    }
    __syncthreads();

    // A fragment: lane<16 holds row, K 0..15; lane>=16: K 16..31
    v16bf afrag, bfrag0, bfrag1;
    const float* ap = A + arow * (unsigned)lda + (unsigned)(kb + ahalf);
    #pragma unroll
    for (int i = 0; i < 8; ++i) {
      afrag[i]     = bits2bf(f2bf_bits(ap[i]));
      afrag[i + 8] = bits2bf(f2bf_bits(ap[i + 8]));
    }
    // B fragments: lane<16 -> column, K 0..15; lane>=16 -> K 16..31
    #pragma unroll
    for (int i = 0; i < 16; ++i) {
      bfrag0[i] = bits2bf(ldsB[bcol][bko + i]);
      bfrag1[i] = bits2bf(ldsB[16 + bcol][bko + i]);
    }

    acc0 = __builtin_amdgcn_wmma_f32_16x16x32_bf16(false, afrag, false, bfrag0,
                                                   (short)0, acc0, false, false);
    acc1 = __builtin_amdgcn_wmma_f32_16x16x32_bf16(false, afrag, false, bfrag1,
                                                   (short)0, acc1, false, false);
  }

  // D layout: lane<16 -> col=lane, rows mrow0+0..7 ; lane>=16 -> rows mrow0+8..15
  const unsigned ccol0 = (unsigned)(col0 + bcol);
  const unsigned crow0 = (unsigned)(mrow0 + ((lane < 16) ? 0 : 8));
  #pragma unroll
  for (int j = 0; j < 8; ++j) {
    unsigned base = (crow0 + (unsigned)j) * (unsigned)ldc;
    C[base + ccol0]      = acc0[j];
    C[base + ccol0 + 16] = acc1[j];
  }
}

// ---------------------------------------------------------------------------
// Edge in-degree counts (mean divisor)
// ---------------------------------------------------------------------------
__global__ __launch_bounds__(256)
void count_edges(const int* __restrict__ dst, int* __restrict__ cnt, int E)
{
  int e = blockIdx.x * blockDim.x + threadIdx.x;
  if (e < E) atomicAdd(&cnt[dst[e]], 1);
}

// ---------------------------------------------------------------------------
// Scatter messages: sums[dst] += w_e * T[src]. One wave per edge, float4/lane.
// ---------------------------------------------------------------------------
__global__ __launch_bounds__(256)
void scatter_mean_msgs(const float* __restrict__ T, const int* __restrict__ src,
                       const int* __restrict__ dst, const float* __restrict__ w,
                       float* __restrict__ sums, int E)
{
  int t = blockIdx.x * 256 + threadIdx.x;
  int e = t >> 5;
  int lane = t & 31;
  if (e >= E) return;
  float wt = w ? w[e] : 1.0f;
  if (wt == 0.0f) return;
  unsigned s = (unsigned)src[e], d = (unsigned)dst[e];
  const float4 v = ((const float4*)(T + s * (unsigned)HD))[lane];
  float* o = sums + d * (unsigned)HD + (unsigned)(lane * 4);
  atomicAdd(o + 0, v.x * wt);
  atomicAdd(o + 1, v.y * wt);
  atomicAdd(o + 2, v.z * wt);
  atomicAdd(o + 3, v.w * wt);
}

// ---------------------------------------------------------------------------
// h = relu(self + s0/max(c0,1) [+ s1/max(c1,1)])
// ---------------------------------------------------------------------------
__global__ __launch_bounds__(256)
void combine_relu(const float* __restrict__ selfT,
                  const float* __restrict__ s0, const int* __restrict__ c0,
                  const float* __restrict__ s1, const int* __restrict__ c1,
                  float* __restrict__ outh, int n)
{
  int idx = blockIdx.x * 256 + threadIdx.x;
  if (idx >= n * HD) return;
  int row = idx >> 7;   // HD == 128
  float v = selfT[idx];
  if (s0) v += s0[idx] / fmaxf((float)c0[row], 1.0f);
  if (s1) v += s1[idx] / fmaxf((float)c1[row], 1.0f);
  outh[idx] = fmaxf(v, 0.0f);
}

// ---------------------------------------------------------------------------
// Edge predictor (column 0 only) + scatter-add into dense A (dup coalesce=sum)
// pred = relu(P1[src] + P2[dst] + b1) . W2[:,0] + b2[0]
// ---------------------------------------------------------------------------
__global__ __launch_bounds__(256)
void edge_pred_scatter(const float* __restrict__ P1, const float* __restrict__ P2,
                       const float* __restrict__ b1, const float* __restrict__ W2,
                       const float* __restrict__ b2,
                       const int* __restrict__ src, const int* __restrict__ dst,
                       float* __restrict__ A, int E, int n)
{
  __shared__ __align__(16) float sb[256];
  __shared__ __align__(16) float sw[256];
  for (int i = threadIdx.x; i < 256; i += 256) { sb[i] = b1[i]; sw[i] = W2[2 * i]; }
  __syncthreads();
  int e = blockIdx.x * 256 + threadIdx.x;
  if (e >= E) return;
  unsigned s = (unsigned)src[e], d = (unsigned)dst[e];
  const float4* p1 = (const float4*)(P1 + s * 256u);
  const float4* p2 = (const float4*)(P2 + d * 256u);
  const float4* b4 = (const float4*)sb;
  const float4* w4 = (const float4*)sw;
  float acc = b2[0];
  #pragma unroll 4
  for (int k = 0; k < 64; ++k) {
    float4 x1 = p1[k], x2 = p2[k], bb = b4[k], ww = w4[k];
    acc += fmaxf(x1.x + x2.x + bb.x, 0.f) * ww.x;
    acc += fmaxf(x1.y + x2.y + bb.y, 0.f) * ww.y;
    acc += fmaxf(x1.z + x2.z + bb.z, 0.f) * ww.z;
    acc += fmaxf(x1.w + x2.w + bb.w, 0.f) * ww.w;
  }
  atomicAdd(&A[(size_t)s * (unsigned)n + d], acc);
}

// ---------------------------------------------------------------------------
// Row-wise gumbel top-5: y = (A>0) ? A + g : -1e9 ; record top-5 indices.
// Block of 256 per row; per-thread top-5 then serial merge. Ties -> low index
// (matches jax.lax.top_k stability).
// ---------------------------------------------------------------------------
static __device__ __forceinline__ float gumbel_noise(unsigned row, unsigned col,
                                                     unsigned seed)
{
  unsigned x = row * 0x9E3779B1u + col * 0x85EBCA77u + seed * 0xC2B2AE3Du + 0x27D4EB2Fu;
  x ^= x >> 16; x *= 0x7FEB352Du; x ^= x >> 15; x *= 0x846CA68Bu; x ^= x >> 16;
  float u = ((float)x + 1.0f) * 2.3283064e-10f;
  u = fminf(fmaxf(u, 1.0e-10f), 0.9999999f);
  return -logf(-logf(u));
}

__global__ __launch_bounds__(256)
void row_topk(const float* __restrict__ A, int n, int* __restrict__ top5,
              unsigned seed)
{
  __shared__ float sv[256 * 5];
  __shared__ int   si[256 * 5];
  const int row = blockIdx.x;
  const float* a = A + (size_t)row * (unsigned)n;

  float bv[5]; int bi[5];
  #pragma unroll
  for (int j = 0; j < 5; ++j) { bv[j] = -3.0e38f; bi[j] = 0x7fffffff; }

  for (int c = threadIdx.x; c < n; c += 256) {
    float av = a[c];
    float y = (av > 0.0f) ? (av + gumbel_noise((unsigned)row, (unsigned)c, seed))
                          : -1.0e9f;
    if (y > bv[4] || (y == bv[4] && c < bi[4])) {
      bv[4] = y; bi[4] = c;
      #pragma unroll
      for (int j = 4; j > 0; --j) {
        if (bv[j] > bv[j - 1] || (bv[j] == bv[j - 1] && bi[j] < bi[j - 1])) {
          float tv = bv[j]; bv[j] = bv[j - 1]; bv[j - 1] = tv;
          int ti = bi[j]; bi[j] = bi[j - 1]; bi[j - 1] = ti;
        }
      }
    }
  }
  #pragma unroll
  for (int j = 0; j < 5; ++j) {
    sv[threadIdx.x * 5 + j] = bv[j];
    si[threadIdx.x * 5 + j] = bi[j];
  }
  __syncthreads();
  if (threadIdx.x == 0) {
    float fv[5]; int fi[5];
    for (int j = 0; j < 5; ++j) { fv[j] = -3.0e38f; fi[j] = 0x7fffffff; }
    for (int t = 0; t < 256 * 5; ++t) {
      float y = sv[t]; int c = si[t];
      if (y > fv[4] || (y == fv[4] && c < fi[4])) {
        fv[4] = y; fi[4] = c;
        for (int j = 4; j > 0; --j) {
          if (fv[j] > fv[j - 1] || (fv[j] == fv[j - 1] && fi[j] < fi[j - 1])) {
            float tv = fv[j]; fv[j] = fv[j - 1]; fv[j - 1] = tv;
            int ti = fi[j]; fi[j] = fi[j - 1]; fi[j - 1] = ti;
          }
        }
      }
    }
    for (int j = 0; j < 5; ++j) top5[row * 5 + j] = fi[j];
  }
}

// ---------------------------------------------------------------------------
// w[e] = 1 if dst[e] in top5[src[e]] else 0   (hard ST mask forward value)
// ---------------------------------------------------------------------------
__global__ __launch_bounds__(256)
void edge_weight(const int* __restrict__ src, const int* __restrict__ dst,
                 const int* __restrict__ top5, float* __restrict__ w, int E)
{
  int e = blockIdx.x * 256 + threadIdx.x;
  if (e >= E) return;
  int s = src[e], d = dst[e];
  float r = 0.0f;
  #pragma unroll
  for (int j = 0; j < 5; ++j)
    if (top5[s * 5 + j] == d) r = 1.0f;
  w[e] = r;
}

// ---------------------------------------------------------------------------
// y[0..4] = h2_paper[index] @ Wc + bc
// ---------------------------------------------------------------------------
__global__ void classify(const float* __restrict__ h, const int* __restrict__ index,
                         const float* __restrict__ Wc, const float* __restrict__ bc,
                         float* __restrict__ out)
{
  int j = threadIdx.x;
  if (j >= 5) return;
  const float* row = h + (unsigned)index[0] * (unsigned)HD;
  float acc = bc[j];
  for (int k = 0; k < HD; ++k) acc += row[k] * Wc[k * 5 + j];
  out[j] = acc;
}

// ---------------------------------------------------------------------------
extern "C" void kernel_launch(void* const* d_in, const int* in_sizes, int n_in,
                              void* d_out, int out_size, void* d_ws, size_t ws_size,
                              hipStream_t stream)
{
  (void)n_in; (void)out_size; (void)ws_size;

  const float* x_p = (const float*)d_in[0];
  const float* x_a = (const float*)d_in[1];
  const int* ei_pp = (const int*)d_in[2];
  const int* ei_aa = (const int*)d_in[3];
  const int* ei_pa = (const int*)d_in[4];
  const int* ei_ap = (const int*)d_in[5];
  const int* index = (const int*)d_in[8];

  const float* Wself_p0 = (const float*)d_in[9];
  const float* Wself_p1 = (const float*)d_in[10];
  const float* Wself_a0 = (const float*)d_in[11];
  const float* Wself_a1 = (const float*)d_in[12];
  const float* Wpp0 = (const float*)d_in[13];
  const float* Wpp1 = (const float*)d_in[14];
  const float* Waa0 = (const float*)d_in[15];
  const float* Waa1 = (const float*)d_in[16];
  const float* Wpa0 = (const float*)d_in[17];
  const float* Wpa1 = (const float*)d_in[18];
  const float* Wap0 = (const float*)d_in[19];
  const float* Wap1 = (const float*)d_in[20];
  const float* Wep1_pp = (const float*)d_in[21];
  const float* bep1_pp = (const float*)d_in[22];
  const float* Wep2_pp = (const float*)d_in[23];
  const float* bep2_pp = (const float*)d_in[24];
  const float* Wep1_aa = (const float*)d_in[25];
  const float* bep1_aa = (const float*)d_in[26];
  const float* Wep2_aa = (const float*)d_in[27];
  const float* bep2_aa = (const float*)d_in[28];
  const float* Wc = (const float*)d_in[29];
  const float* bc = (const float*)d_in[30];

  const int Epp = in_sizes[2] / 2, Eaa = in_sizes[3] / 2;
  const int Epa = in_sizes[4] / 2, Eap = in_sizes[5] / 2;
  const int *src_pp = ei_pp, *dst_pp = ei_pp + Epp;
  const int *src_aa = ei_aa, *dst_aa = ei_aa + Eaa;
  const int *src_pa = ei_pa, *dst_pa = ei_pa + Epa;
  const int *src_ap = ei_ap, *dst_ap = ei_ap + Eap;

  float* out  = (float*)d_out;
  float* w_pp = out + 5;
  float* w_aa = out + 5 + Epp;

  // ------------------ workspace bump allocator ------------------
  size_t off = 0;
  auto wsalloc = [&](size_t bytes) -> void* {
    void* p = (void*)((char*)d_ws + off);
    off += (bytes + 255) & ~(size_t)255;
    return p;
  };
  const size_t NPH = (size_t)NPAP * HD * 4, NAH = (size_t)NAUT * HD * 4;

  int* cnt_pp = (int*)wsalloc(NPAP * 4);
  int* cnt_aa = (int*)wsalloc(NAUT * 4);
  int* cnt_ap = (int*)wsalloc(NPAP * 4);   // dst = paper
  int* cnt_pa = (int*)wsalloc(NAUT * 4);   // dst = author
  float* Tself_p0 = (float*)wsalloc(NPH);
  float* Tpp0     = (float*)wsalloc(NPH);
  float* Tself_a0 = (float*)wsalloc(NAH);
  float* Taa0     = (float*)wsalloc(NAH);
  float* sumP0 = (float*)wsalloc(NPH);
  float* sumP1 = (float*)wsalloc(NPH);
  float* sumA0 = (float*)wsalloc(NAH);
  float* sumA1 = (float*)wsalloc(NAH);
  float* h1l0_p = (float*)wsalloc(NPH);
  float* h1l0_a = (float*)wsalloc(NAH);
  float* Tbuf_p1 = (float*)wsalloc(NPH);   // reused: self/msg GEMM outs, both GNNs L1
  float* Tbuf_p2 = (float*)wsalloc(NPH);
  float* Tbuf_a1 = (float*)wsalloc(NAH);
  float* Tbuf_a2 = (float*)wsalloc(NAH);
  float* h1_p = (float*)wsalloc(NPH);
  float* h1_a = (float*)wsalloc(NAH);
  float* P1_pp = (float*)wsalloc((size_t)NPAP * 256 * 4);
  float* P2_pp = (float*)wsalloc((size_t)NPAP * 256 * 4);
  float* P1_aa = (float*)wsalloc((size_t)NAUT * 256 * 4);
  float* P2_aa = (float*)wsalloc((size_t)NAUT * 256 * 4);
  int* top5_pp = (int*)wsalloc(NPAP * 5 * 4);
  int* top5_aa = (int*)wsalloc(NAUT * 5 * 4);
  float* A_pp = (float*)wsalloc((size_t)NPAP * NPAP * 4);   // 64 MB
  float* A_aa = (float*)wsalloc((size_t)NAUT * NAUT * 4);   // 16 MB
  // GNN2 temporaries alias into A_pp (prune is finished before GNN2 starts)
  float* Tap0   = A_pp;
  float* Tpa0   = Tap0 + (size_t)NAUT * HD;
  float* h2l0_p = Tpa0 + (size_t)NPAP * HD;
  float* h2l0_a = h2l0_p + (size_t)NPAP * HD;
  float* h2_p   = h2l0_a + (size_t)NAUT * HD;

  auto GEMM = [&](const float* A, const float* B, float* C,
                  int M, int N, int lda, int ldb, int ldc) {
    dim3 g(N / 32, M / 64);
    gemm_bf16_wmma<<<g, 128, 0, stream>>>(A, B, C, lda, ldb, ldc);
  };
  auto SCATTER = [&](const float* T, const int* s, const int* d, const float* w,
                     float* sums, int E) {
    scatter_mean_msgs<<<(E * 32 + 255) / 256, 256, 0, stream>>>(T, s, d, w, sums, E);
  };

  // ------------------ edge counts (fixed for all layers) ------------------
  hipMemsetAsync(cnt_pp, 0, NPAP * 4, stream);
  hipMemsetAsync(cnt_aa, 0, NAUT * 4, stream);
  hipMemsetAsync(cnt_ap, 0, NPAP * 4, stream);
  hipMemsetAsync(cnt_pa, 0, NAUT * 4, stream);
  count_edges<<<(Epp + 255) / 256, 256, 0, stream>>>(dst_pp, cnt_pp, Epp);
  count_edges<<<(Eaa + 255) / 256, 256, 0, stream>>>(dst_aa, cnt_aa, Eaa);
  count_edges<<<(Eap + 255) / 256, 256, 0, stream>>>(dst_ap, cnt_ap, Eap);
  count_edges<<<(Epa + 255) / 256, 256, 0, stream>>>(dst_pa, cnt_pa, Epa);

  // ------------------ GNN1 (single=True) layer 0 ------------------
  GEMM(x_p, Wself_p0, Tself_p0, NPAP, HD, HD, HD, HD);
  GEMM(x_p, Wpp0,     Tpp0,     NPAP, HD, HD, HD, HD);
  GEMM(x_a, Wself_a0, Tself_a0, NAUT, HD, HD, HD, HD);
  GEMM(x_a, Waa0,     Taa0,     NAUT, HD, HD, HD, HD);
  hipMemsetAsync(sumP0, 0, NPH, stream);
  hipMemsetAsync(sumA0, 0, NAH, stream);
  SCATTER(Tpp0, src_pp, dst_pp, nullptr, sumP0, Epp);
  SCATTER(Taa0, src_aa, dst_aa, nullptr, sumA0, Eaa);
  combine_relu<<<NPAP * HD / 256, 256, 0, stream>>>(Tself_p0, sumP0, cnt_pp,
                                                    nullptr, nullptr, h1l0_p, NPAP);
  combine_relu<<<NAUT * HD / 256, 256, 0, stream>>>(Tself_a0, sumA0, cnt_aa,
                                                    nullptr, nullptr, h1l0_a, NAUT);
  // ------------------ GNN1 layer 1 ------------------
  GEMM(h1l0_p, Wself_p1, Tbuf_p1, NPAP, HD, HD, HD, HD);
  GEMM(h1l0_p, Wpp1,     Tbuf_p2, NPAP, HD, HD, HD, HD);
  GEMM(h1l0_a, Wself_a1, Tbuf_a1, NAUT, HD, HD, HD, HD);
  GEMM(h1l0_a, Waa1,     Tbuf_a2, NAUT, HD, HD, HD, HD);
  hipMemsetAsync(sumP0, 0, NPH, stream);
  hipMemsetAsync(sumA0, 0, NAH, stream);
  SCATTER(Tbuf_p2, src_pp, dst_pp, nullptr, sumP0, Epp);
  SCATTER(Tbuf_a2, src_aa, dst_aa, nullptr, sumA0, Eaa);
  combine_relu<<<NPAP * HD / 256, 256, 0, stream>>>(Tbuf_p1, sumP0, cnt_pp,
                                                    nullptr, nullptr, h1_p, NPAP);
  combine_relu<<<NAUT * HD / 256, 256, 0, stream>>>(Tbuf_a1, sumA0, cnt_aa,
                                                    nullptr, nullptr, h1_a, NAUT);

  // ------------------ edge predictor + prune ------------------
  // concat(h_src,h_dst)@Wep1 = h@Wep1[:H,:] (gather src) + h@Wep1[H:,:] (gather dst)
  GEMM(h1_p, Wep1_pp,            P1_pp, NPAP, 256, HD, 256, 256);
  GEMM(h1_p, Wep1_pp + HD * 256, P2_pp, NPAP, 256, HD, 256, 256);
  GEMM(h1_a, Wep1_aa,            P1_aa, NAUT, 256, HD, 256, 256);
  GEMM(h1_a, Wep1_aa + HD * 256, P2_aa, NAUT, 256, HD, 256, 256);
  hipMemsetAsync(A_pp, 0, (size_t)NPAP * NPAP * 4, stream);
  hipMemsetAsync(A_aa, 0, (size_t)NAUT * NAUT * 4, stream);
  edge_pred_scatter<<<(Epp + 255) / 256, 256, 0, stream>>>(
      P1_pp, P2_pp, bep1_pp, Wep2_pp, bep2_pp, src_pp, dst_pp, A_pp, Epp, NPAP);
  edge_pred_scatter<<<(Eaa + 255) / 256, 256, 0, stream>>>(
      P1_aa, P2_aa, bep1_aa, Wep2_aa, bep2_aa, src_aa, dst_aa, A_aa, Eaa, NAUT);
  row_topk<<<NPAP, 256, 0, stream>>>(A_pp, NPAP, top5_pp, 0x2A5F3C19u);
  row_topk<<<NAUT, 256, 0, stream>>>(A_aa, NAUT, top5_aa, 0x7B11D4E5u);
  edge_weight<<<(Epp + 255) / 256, 256, 0, stream>>>(src_pp, dst_pp, top5_pp, w_pp, Epp);
  edge_weight<<<(Eaa + 255) / 256, 256, 0, stream>>>(src_aa, dst_aa, top5_aa, w_aa, Eaa);

  // ------------------ GNN2 (all relations, edge_attn on pp/aa) layer 0 ----
  // Layer-0 inputs are x again -> reuse Tself_p0/Tpp0/Tself_a0/Taa0.
  GEMM(x_a, Wap0, Tap0, NAUT, HD, HD, HD, HD);
  GEMM(x_p, Wpa0, Tpa0, NPAP, HD, HD, HD, HD);
  hipMemsetAsync(sumP0, 0, NPH, stream);
  hipMemsetAsync(sumP1, 0, NPH, stream);
  hipMemsetAsync(sumA0, 0, NAH, stream);
  hipMemsetAsync(sumA1, 0, NAH, stream);
  SCATTER(Tpp0, src_pp, dst_pp, w_pp,    sumP0, Epp);
  SCATTER(Tap0, src_ap, dst_ap, nullptr, sumP1, Eap);
  SCATTER(Taa0, src_aa, dst_aa, w_aa,    sumA0, Eaa);
  SCATTER(Tpa0, src_pa, dst_pa, nullptr, sumA1, Epa);
  combine_relu<<<NPAP * HD / 256, 256, 0, stream>>>(Tself_p0, sumP0, cnt_pp,
                                                    sumP1, cnt_ap, h2l0_p, NPAP);
  combine_relu<<<NAUT * HD / 256, 256, 0, stream>>>(Tself_a0, sumA0, cnt_aa,
                                                    sumA1, cnt_pa, h2l0_a, NAUT);
  // ------------------ GNN2 layer 1 (paper side only feeds the output) ------
  GEMM(h2l0_p, Wself_p1, Tbuf_p1, NPAP, HD, HD, HD, HD);
  GEMM(h2l0_p, Wpp1,     Tbuf_p2, NPAP, HD, HD, HD, HD);
  GEMM(h2l0_a, Wap1,     Tbuf_a1, NAUT, HD, HD, HD, HD);
  hipMemsetAsync(sumP0, 0, NPH, stream);
  hipMemsetAsync(sumP1, 0, NPH, stream);
  SCATTER(Tbuf_p2, src_pp, dst_pp, w_pp,    sumP0, Epp);
  SCATTER(Tbuf_a1, src_ap, dst_ap, nullptr, sumP1, Eap);
  combine_relu<<<NPAP * HD / 256, 256, 0, stream>>>(Tbuf_p1, sumP0, cnt_pp,
                                                    sumP1, cnt_ap, h2_p, NPAP);

  // ------------------ classifier head ------------------
  classify<<<1, 32, 0, stream>>>(h2_p, index, Wc, bc, out);
}
